// Llama_head_62191126446185
// MI455X (gfx1250) — matrix operand order repaired and checked by reading, not compiled, not run on hardware
//
#include <hip/hip_runtime.h>
#include <hip/hip_bf16.h>

typedef __attribute__((ext_vector_type(16))) __bf16 v16bf;
typedef __attribute__((ext_vector_type(8)))  __bf16 v8bf;
typedef __attribute__((ext_vector_type(4)))  __bf16 v4bf;
typedef __attribute__((ext_vector_type(8)))  float  v8f;
typedef __attribute__((ext_vector_type(4)))  unsigned int v4u;
typedef __attribute__((ext_vector_type(8)))  int    v8i;
typedef __attribute__((ext_vector_type(4)))  int    v4i;

#define NROWS 32768
#define D 4096
#define H 256
#define O 64
#define ROWS 64      // rows per workgroup
#define KC 64        // K chunk staged in LDS
#define NCHUNK (D / KC)

#if defined(__HIP_DEVICE_COMPILE__) && __has_builtin(__builtin_amdgcn_tensor_load_to_lds)
#define USE_TDM 1
#else
#define USE_TDM 0
#endif

// ---------------------------------------------------------------------------
// One-time weight prep: fp32 [K][N] row-major  ->  bf16 [N][K] (transposed).
// ---------------------------------------------------------------------------
__global__ __launch_bounds__(256) void transpose_to_bf16(
    const float* __restrict__ src, __bf16* __restrict__ dst, int K, int Ncols)
{
    int idx = blockIdx.x * 256 + threadIdx.x;
    if (idx >= K * Ncols) return;
    int k = idx / Ncols;
    int n = idx - k * Ncols;
    dst[(size_t)n * K + k] = (__bf16)src[(size_t)k * Ncols + n];
}

// ---------------------------------------------------------------------------
// TDM: DMA a 2D bf16 tile [tile_d1 rows x tile_d0 elems] from a row-major
// [tensor_d1][tensor_d0] tensor (row stride = stride0 elems) into LDS,
// packed contiguously.  D# layout per CDNA5 ISA ch.8 (group0/group1; groups
// 2/3 zero for a 2D tensor).  Tracked by TENSORcnt.
// ---------------------------------------------------------------------------
#if USE_TDM
__device__ __forceinline__ void tdm_load_2d_bf16(
    unsigned lds_off, const void* gaddr,
    unsigned tensor_d0, unsigned tensor_d1,
    unsigned tile_d0, unsigned tile_d1, unsigned stride0)
{
    unsigned long long ga = (unsigned long long)gaddr;
    v4u g0;
    g0[0] = 1u;                                             // count=1 (valid), user mode
    g0[1] = lds_off;                                        // lds_addr (bytes)
    g0[2] = (unsigned)ga;                                   // global_addr[31:0]
    g0[3] = (unsigned)((ga >> 32) & 0x1FFFFFFu) | (2u << 30); // addr[56:32] | type=2
    v8i g1;
    g1[0] = (int)(1u << 16);                                // data_size=1 (2B); wg_mask=0
    g1[1] = (int)((tensor_d0 & 0xFFFFu) << 16);             // tensor_dim0[15:0] @bit48
    g1[2] = (int)(((tensor_d0 >> 16) & 0xFFFFu) |
                  ((tensor_d1 & 0xFFFFu) << 16));           // dim0[31:16] | dim1[15:0]
    g1[3] = (int)(((tensor_d1 >> 16) & 0xFFFFu) |
                  ((tile_d0 & 0xFFFFu) << 16));             // dim1[31:16] | tile_dim0
    g1[4] = (int)(tile_d1 & 0xFFFFu);                       // tile_dim1; tile_dim2=0
    g1[5] = (int)stride0;                                   // tensor_dim0_stride[31:0]
    g1[6] = 0;                                              // stride0[47:32]|stride1 lo
    g1[7] = 0;
    v4i z4 = {0, 0, 0, 0};
#if __clang_major__ >= 23
    v8i z8 = {0, 0, 0, 0, 0, 0, 0, 0};
    __builtin_amdgcn_tensor_load_to_lds(g0, g1, z4, z4, z8, 0);
#else
    __builtin_amdgcn_tensor_load_to_lds(g0, g1, z4, z4, 0);
#endif
}
#endif

__device__ __forceinline__ unsigned lds_byte_off(const void* p)
{
    // LDS generic addresses carry the LDS byte offset in the low 32 bits.
    return (unsigned)(unsigned long long)(uintptr_t)p;
}

// ---------------------------------------------------------------------------
// Fused: u = mlp(xu), v = mlp(xi), out = sigmoid(rowdot(u, v))
// 512 blocks x 256 threads (8 wave32), 64 rows per block.
// Layer 1: each wave owns a 16x128 C tile (8 wmma accum tiles); W1^T tiles
// double-buffered in LDS via TDM prefetch.  Layer 2: waves 0-3, 16x64 each.
// ---------------------------------------------------------------------------
__global__ __launch_bounds__(256) void fused_mlp_dot(
    const float*  __restrict__ xu,   const float*  __restrict__ xi,
    const __bf16* __restrict__ w1Tu, const __bf16* __restrict__ w1Ti,
    const __bf16* __restrict__ w2Tu, const __bf16* __restrict__ w2Ti,
    const float*  __restrict__ b1u,  const float*  __restrict__ b1i,
    const float*  __restrict__ b2u,  const float*  __restrict__ b2i,
    float* __restrict__ out)
{
    __shared__ __bf16 xb[ROWS * KC];        //  8 KB: x tile [row][k] bf16
    __shared__ __bf16 wbuf[2 * H * KC];     // 64 KB: double-buffered W1^T tile [n][k]
                                            //        (buffer 0 reused for W2^T [n][k])
    __shared__ __bf16 hb[ROWS * H];         // 32 KB: hidden activations [row][k]
    __shared__ float  ub[ROWS * O];         // 16 KB
    __shared__ float  vb[ROWS * O];         // 16 KB

    const int tid   = threadIdx.x;
    const int lane  = tid & 31;
    const int wave  = tid >> 5;
    const int wr    = wave & 3;    // row group (4 x 16 rows)
    const int wc    = wave >> 2;   // col half  (2 x 128 cols)
    const int ln15  = lane & 15;
    const int lhalf = lane >> 4;
    const int row0  = blockIdx.x * ROWS;

    for (int path = 0; path < 2; ++path) {
        const float*  x   = path ? xi   : xu;
        const __bf16* w1T = path ? w1Ti : w1Tu;
        const __bf16* w2T = path ? w2Ti : w2Tu;
        const float*  b1  = path ? b1i  : b1u;
        const float*  b2  = path ? b2i  : b2u;
        float*        res = path ? vb   : ub;

        v8f acc[8] = {};   // 16 rows x 128 cols per wave

#if USE_TDM
        // Prefetch chunk 0's W1^T tile into buffer 0.
        if (wave == 0)
            tdm_load_2d_bf16(lds_byte_off(&wbuf[0]), w1T, D, H, KC, H, D);
#endif

        for (int ic = 0; ic < NCHUNK; ++ic) {
            const int kc  = ic * KC;
            const int cur = ic & 1;
            __bf16* wb = wbuf + cur * (H * KC);

            // ---- stage x tile: fp32 global -> bf16 LDS ----
            #pragma unroll
            for (int i = 0; i < 4; ++i) {
                int f = tid + i * 256;           // 1024 float4 per tile
                int r = f >> 4;
                int c = (f & 15) << 2;
                const float* gp = x + (size_t)(row0 + r) * D + kc + c;
                float4 v = *(const float4*)gp;
                if (kc + KC < D) __builtin_prefetch(gp + KC, 0, 1);
                v4bf p = { (__bf16)v.x, (__bf16)v.y, (__bf16)v.z, (__bf16)v.w };
                *(v4bf*)(xb + r * KC + c) = p;
            }

#if USE_TDM
            if (wave == 0) {
                if (ic + 1 < NCHUNK) {
                    // Prefetch next chunk into the other buffer, then ensure
                    // the current chunk's DMA (issued earlier) has landed.
                    tdm_load_2d_bf16(lds_byte_off(&wbuf[(cur ^ 1) * (H * KC)]),
                                     w1T + kc + KC, D, H, KC, H, D);
                    __builtin_amdgcn_s_wait_tensorcnt((short)1);
                } else {
                    __builtin_amdgcn_s_wait_tensorcnt((short)0);
                }
            }
#else
            {
                const uint4* src = (const uint4*)(w1T + (size_t)tid * D + kc);
                uint4* dst = (uint4*)(wb + tid * KC);
                #pragma unroll
                for (int i = 0; i < 8; ++i) dst[i] = src[i];
            }
#endif
            __syncthreads();

            // ---- 2 K-steps of 32: preload 8 B frags, then 8 WMMAs ----
            #pragma unroll
            for (int ks = 0; ks < KC; ks += 32) {
                v16bf bfr[8];
                #pragma unroll
                for (int t = 0; t < 8; ++t)
                    bfr[t] = __builtin_shufflevector(
                        *(const v8bf*)(wb + (wc * 128 + t * 16 + ln15) * KC + ks + lhalf * 8),
                        *(const v8bf*)(wb + (wc * 128 + t * 16 + ln15) * KC + ks + lhalf * 8 + 16),
                        0,1,2,3,4,5,6,7,8,9,10,11,12,13,14,15);
                const __bf16* ap = xb + (wr * 16 + ln15) * KC + ks + lhalf * 8;
                v8bf alo = *(const v8bf*)ap;
                v8bf ahi = *(const v8bf*)(ap + 16);
                v16bf a = __builtin_shufflevector(alo, ahi,
                        0,1,2,3,4,5,6,7,8,9,10,11,12,13,14,15);
                #pragma unroll
                for (int t = 0; t < 8; ++t)
                    acc[t] = __builtin_amdgcn_wmma_f32_16x16x32_bf16(
                        false, a, false, bfr[t], (short)0, acc[t], false, false);
            }
            __syncthreads();
        }

        // ---- layer-1 epilogue: bias + ReLU, h -> LDS bf16 [row][k] ----
        #pragma unroll
        for (int t = 0; t < 8; ++t) {
            int n = wc * 128 + t * 16 + ln15;
            float bias = b1[n];
            #pragma unroll
            for (int g = 0; g < 8; ++g) {
                int m = lhalf * 8 + g;
                float hv = acc[t][g] + bias;
                hb[(wr * 16 + m) * H + n] = (__bf16)(hv > 0.f ? hv : 0.f);
            }
        }
        // ---- stage W2^T [64][256] bf16 into wbuf[0] ----
        {
            uint4* dst = (uint4*)wbuf;
            const uint4* src = (const uint4*)w2T;
            #pragma unroll
            for (int i = 0; i < 8; ++i) dst[tid * 8 + i] = src[tid * 8 + i];
        }
        __syncthreads();

        // ---- layer 2: 64x256 @ 256x64, waves 0-3 ----
        if (wave < 4) {
            const int r0 = wave * 16;
            v8f acc2[4] = {};
            #pragma unroll
            for (int ks = 0; ks < H; ks += 32) {
                v16bf bfr[4];
                #pragma unroll
                for (int t = 0; t < 4; ++t)
                    bfr[t] = __builtin_shufflevector(
                        *(const v8bf*)(wbuf + (t * 16 + ln15) * H + ks + lhalf * 8),
                        *(const v8bf*)(wbuf + (t * 16 + ln15) * H + ks + lhalf * 8 + 16),
                        0,1,2,3,4,5,6,7,8,9,10,11,12,13,14,15);
                const __bf16* ap = hb + (r0 + ln15) * H + ks + lhalf * 8;
                v8bf alo = *(const v8bf*)ap;
                v8bf ahi = *(const v8bf*)(ap + 16);
                v16bf a = __builtin_shufflevector(alo, ahi,
                        0,1,2,3,4,5,6,7,8,9,10,11,12,13,14,15);
                #pragma unroll
                for (int t = 0; t < 4; ++t)
                    acc2[t] = __builtin_amdgcn_wmma_f32_16x16x32_bf16(
                        false, a, false, bfr[t], (short)0, acc2[t], false, false);
            }
            #pragma unroll
            for (int t = 0; t < 4; ++t) {
                int n = t * 16 + ln15;
                float bias = b2[n];
                #pragma unroll
                for (int g = 0; g < 8; ++g)
                    res[(r0 + lhalf * 8 + g) * O + n] = acc2[t][g] + bias;
            }
        }
        __syncthreads();
    }

    // ---- rowwise dot(u, v) + sigmoid ----
    if (tid < O) {
        const float4* ur = (const float4*)(ub + tid * O);
        const float4* vr = (const float4*)(vb + tid * O);
        float s = 0.f;
        #pragma unroll
        for (int i = 0; i < O / 4; ++i) {
            float4 a = ur[i], b = vr[i];
            s += a.x * b.x + a.y * b.y + a.z * b.z + a.w * b.w;
        }
        out[row0 + tid] = 1.0f / (1.0f + __expf(-s));
    }
}

// ---------------------------------------------------------------------------
extern "C" void kernel_launch(void* const* d_in, const int* in_sizes, int n_in,
                              void* d_out, int out_size, void* d_ws, size_t ws_size,
                              hipStream_t stream)
{
    const float* xu  = (const float*)d_in[0];
    const float* xi  = (const float*)d_in[1];
    const float* uw1 = (const float*)d_in[2];
    const float* ub1 = (const float*)d_in[3];
    const float* uw2 = (const float*)d_in[4];
    const float* ub2 = (const float*)d_in[5];
    const float* iw1 = (const float*)d_in[6];
    const float* ib1 = (const float*)d_in[7];
    const float* iw2 = (const float*)d_in[8];
    const float* ib2 = (const float*)d_in[9];
    float* out = (float*)d_out;

    __bf16* ws   = (__bf16*)d_ws;
    __bf16* uw1T = ws;                                   // [H][D]
    __bf16* iw1T = ws + (size_t)D * H;                   // [H][D]
    __bf16* uw2T = ws + (size_t)2 * D * H;               // [O][H]
    __bf16* iw2T = uw2T + (size_t)H * O;                 // [O][H]

    transpose_to_bf16<<<(D * H + 255) / 256, 256, 0, stream>>>(uw1, uw1T, D, H);
    transpose_to_bf16<<<(D * H + 255) / 256, 256, 0, stream>>>(iw1, iw1T, D, H);
    transpose_to_bf16<<<(H * O + 255) / 256, 256, 0, stream>>>(uw2, uw2T, H, O);
    transpose_to_bf16<<<(H * O + 255) / 256, 256, 0, stream>>>(iw2, iw2T, H, O);

    fused_mlp_dot<<<NROWS / ROWS, 256, 0, stream>>>(
        xu, xi, uw1T, iw1T, uw2T, iw2T, ub1, ib1, ub2, ib2, out);
}
